// FC_small_283467841698
// MI455X (gfx1250) — 8-bit matrix operand order repaired and checked by reading, compiled, not run on hardware
//
#include <hip/hip_runtime.h>
#include <cstdint>

typedef int v8i __attribute__((ext_vector_type(8)));
typedef int v4i __attribute__((vector_size(16)));

#define NB     65536
#define IND    768
#define HID    512
#define NOUT   10
#define OUTP   16
#define BN_EPS 1e-5f

#define AS1 __attribute__((address_space(1)))
#define AS3 __attribute__((address_space(3)))

#if defined(__AMDGCN__) && __has_builtin(__builtin_amdgcn_global_load_async_to_lds_b128)
#define HAS_ASYNC_LDS 1
#endif

__device__ __forceinline__ void cp16(const int8_t* __restrict__ g, int8_t* l) {
#if defined(HAS_ASYNC_LDS)
    __builtin_amdgcn_global_load_async_to_lds_b128(
        (AS1 v4i*)(uintptr_t)g,
        (AS3 v4i*)(uint32_t)(uintptr_t)l,
        0, 0);
#else
    *(int4*)l = *(const int4*)g;
#endif
}

__device__ __forceinline__ void async_fence() {
#if defined(HAS_ASYNC_LDS)
#if __has_builtin(__builtin_amdgcn_s_wait_asynccnt)
    __builtin_amdgcn_s_wait_asynccnt(0);
#else
    asm volatile("s_wait_asynccnt 0" ::: "memory");
#endif
#endif
}

__global__ void k_bin4(const float* __restrict__ in, int8_t* __restrict__ out, int n4) {
    int i = blockIdx.x * blockDim.x + threadIdx.x;
    if (i >= n4) return;
    float4 v = ((const float4*)in)[i];
    char4 o;
    o.x = v.x > 0.f ? 1 : (v.x < 0.f ? -1 : 0);
    o.y = v.y > 0.f ? 1 : (v.y < 0.f ? -1 : 0);
    o.z = v.z > 0.f ? 1 : (v.z < 0.f ? -1 : 0);
    o.w = v.w > 0.f ? 1 : (v.w < 0.f ? -1 : 0);
    ((char4*)out)[i] = o;
}

__global__ void k_bin_w2(const float* __restrict__ W2, int8_t* __restrict__ w2q) {
    int i = blockIdx.x * blockDim.x + threadIdx.x;
    if (i >= OUTP * HID) return;
    int n = i >> 9, k = i & 511;
    int8_t v = 0;
    if (n < NOUT) {
        float f = W2[n * HID + k];
        v = f > 0.f ? 1 : (f < 0.f ? -1 : 0);
    }
    w2q[i] = v;
}

__global__ void __launch_bounds__(256) k_gemm1(const int8_t* __restrict__ xq,
                                               const int8_t* __restrict__ w1q,
                                               int16_t* __restrict__ h16) {
    __shared__ int8_t As[2][128 * 80];
    __shared__ int8_t Bs[2][64 * 80];
    const int bid = blockIdx.x;
    const int m0 = (bid >> 3) * 128;
    const int n0 = (bid & 7) * 64;
    const int t = threadIdx.x;
    const int wave = t >> 5, lane = t & 31;
    const int lm = lane & 15, lh = lane >> 4;

    const int ar0 = (t * 2) >> 2,       ak0 = ((t * 2) & 3) << 4;
    const int ar1 = (t * 2 + 1) >> 2,   ak1 = ((t * 2 + 1) & 3) << 4;
    const int br  = t >> 2,             bk  = (t & 3) << 4;
    const int8_t* gA0 = xq + (size_t)(m0 + ar0) * IND + ak0;
    const int8_t* gA1 = xq + (size_t)(m0 + ar1) * IND + ak1;
    const int8_t* gB  = w1q + (size_t)(n0 + br) * IND + bk;

    v8i acc[4];
#pragma unroll
    for (int i = 0; i < 4; ++i) acc[i] = 0;

    cp16(gA0, &As[0][ar0 * 80 + ak0]);
    cp16(gA1, &As[0][ar1 * 80 + ak1]);
    cp16(gB, &Bs[0][br * 80 + bk]);
    async_fence();
    __syncthreads();

#pragma unroll
    for (int kk = 0; kk < IND / 64; ++kk) {
        const int cur = kk & 1;
        if (kk + 1 < IND / 64) {
            const int kb = (kk + 1) * 64;
            cp16(gA0 + kb, &As[cur ^ 1][ar0 * 80 + ak0]);
            cp16(gA1 + kb, &As[cur ^ 1][ar1 * 80 + ak1]);
            cp16(gB + kb, &Bs[cur ^ 1][br * 80 + bk]);
        }

        const int* Asw = (const int*)As[cur];
        const int* Bsw = (const int*)Bs[cur];
        v8i afrag;
        const int am = wave * 16 + lm;
#pragma unroll
        for (int i = 0; i < 8; ++i) {
            int k = ((i >> 1) << 4) + (lh << 3) + ((i & 1) << 2);
            afrag[i] = Asw[(am * 80 + k) >> 2];
        }
#pragma unroll
        for (int ns = 0; ns < 4; ++ns) {
            v8i bfrag;
            const int bn = ns * 16 + lm;
#pragma unroll
            for (int i = 0; i < 8; ++i) {
                int k = ((i >> 1) << 4) + (lh << 3) + ((i & 1) << 2);
                bfrag[i] = Bsw[(bn * 80 + k) >> 2];
            }
            acc[ns] = __builtin_amdgcn_wmma_i32_16x16x64_iu8(
                true, afrag, true, bfrag, acc[ns], false, false);
        }
        async_fence();
        __syncthreads();
    }

#pragma unroll
    for (int ns = 0; ns < 4; ++ns) {
#pragma unroll
        for (int r = 0; r < 8; ++r) {
            int row = m0 + wave * 16 + (lh ? r + 8 : r);
            int col = n0 + ns * 16 + lm;
            h16[(size_t)row * HID + col] = (int16_t)acc[ns][r];
        }
    }
}

__global__ void __launch_bounds__(256) k_red1(const int16_t* __restrict__ h16,
                                              int* __restrict__ psum,
                                              int* __restrict__ psq) {
    int b = blockIdx.x, t = threadIdx.x;
    int s0 = 0, s1 = 0, q0 = 0, q1 = 0;
    size_t base = (size_t)b * 256 * HID;
    for (int r = 0; r < 256; ++r) {
        int v0 = h16[base + (size_t)r * HID + t];
        int v1 = h16[base + (size_t)r * HID + t + 256];
        s0 += v0; q0 += v0 * v0;
        s1 += v1; q1 += v1 * v1;
    }
    psum[b * HID + t] = s0;       psum[b * HID + t + 256] = s1;
    psq[b * HID + t] = q0;        psq[b * HID + t + 256] = q1;
}

__global__ void k_fin1(const int* __restrict__ psum, const int* __restrict__ psq,
                       const float* __restrict__ g1, const float* __restrict__ b1,
                       float* __restrict__ scale1, float* __restrict__ bias1) {
    int c = blockIdx.x * blockDim.x + threadIdx.x;
    if (c >= HID) return;
    long long s = 0, q = 0;
    for (int b = 0; b < 256; ++b) { s += psum[b * HID + c]; q += psq[b * HID + c]; }
    double mu = (double)s / (double)NB;
    double var = (double)q / (double)NB - mu * mu;
    float inv = rsqrtf((float)var + BN_EPS);
    float sc = g1[c] * inv;
    scale1[c] = sc;
    bias1[c] = b1[c] - (float)mu * sc;
}

__global__ void __launch_bounds__(256) k_gemm2(const int16_t* __restrict__ h16,
                                               const int8_t* __restrict__ w2q,
                                               const float* __restrict__ scale1,
                                               const float* __restrict__ bias1,
                                               float* __restrict__ o_raw,
                                               float* __restrict__ p2sum,
                                               float* __restrict__ p2sq) {
    __shared__ int8_t Bs[OUTP * 512];
    __shared__ float sS[512], sB[512];
    __shared__ float csum[OUTP], csq[OUTP];
    const int t = threadIdx.x, wave = t >> 5, lane = t & 31;
    const int lm = lane & 15, lh = lane >> 4;

    {
        cp16(w2q + t * 16, &Bs[t * 16]);
        cp16(w2q + (t + 256) * 16, &Bs[(t + 256) * 16]);
        sS[t] = scale1[t]; sS[t + 256] = scale1[t + 256];
        sB[t] = bias1[t];  sB[t + 256] = bias1[t + 256];
        if (t < OUTP) { csum[t] = 0.f; csq[t] = 0.f; }
    }
    async_fence();
    __syncthreads();

    const int m = blockIdx.x * 128 + wave * 16 + lm;
    const int* Bsw = (const int*)Bs;
    v8i acc = 0;

#pragma unroll
    for (int kk = 0; kk < HID / 64; ++kk) {
        v8i afrag;
#pragma unroll
        for (int i = 0; i < 8; ++i) {
            int k = kk * 64 + ((i >> 1) << 4) + (lh << 3) + ((i & 1) << 2);
            short4 hv = *(const short4*)(h16 + (size_t)m * HID + k);
            float f0 = (float)hv.x * sS[k]     + sB[k];
            float f1 = (float)hv.y * sS[k + 1] + sB[k + 1];
            float f2 = (float)hv.z * sS[k + 2] + sB[k + 2];
            float f3 = (float)hv.w * sS[k + 3] + sB[k + 3];
            unsigned u0 = (unsigned char)(signed char)(f0 > 0.f ? 1 : (f0 < 0.f ? -1 : 0));
            unsigned u1 = (unsigned char)(signed char)(f1 > 0.f ? 1 : (f1 < 0.f ? -1 : 0));
            unsigned u2 = (unsigned char)(signed char)(f2 > 0.f ? 1 : (f2 < 0.f ? -1 : 0));
            unsigned u3 = (unsigned char)(signed char)(f3 > 0.f ? 1 : (f3 < 0.f ? -1 : 0));
            afrag[i] = (int)(u0 | (u1 << 8) | (u2 << 16) | (u3 << 24));
        }
        v8i bfrag;
#pragma unroll
        for (int i = 0; i < 8; ++i) {
            int k = kk * 64 + ((i >> 1) << 4) + (lh << 3) + ((i & 1) << 2);
            bfrag[i] = Bsw[(lm * 512 + k) >> 2];
        }
        acc = __builtin_amdgcn_wmma_i32_16x16x64_iu8(
            true, afrag, true, bfrag, acc, false, false);
    }

    float rs = 0.f, rq = 0.f;
#pragma unroll
    for (int r = 0; r < 8; ++r) {
        int row = blockIdx.x * 128 + wave * 16 + (lh ? r + 8 : r);
        float v = (float)acc[r];
        o_raw[(size_t)row * OUTP + lm] = v;
        rs += v; rq += v * v;
    }
    rs += __shfl_xor(rs, 16, 32);
    rq += __shfl_xor(rq, 16, 32);
    if (lh == 0) {
        atomicAdd(&csum[lm], rs);
        atomicAdd(&csq[lm], rq);
    }
    __syncthreads();
    if (t < OUTP) {
        p2sum[blockIdx.x * OUTP + t] = csum[t];
        p2sq[blockIdx.x * OUTP + t] = csq[t];
    }
}

__global__ void k_fin2(const float* __restrict__ p2sum, const float* __restrict__ p2sq,
                       const float* __restrict__ g2, const float* __restrict__ b2,
                       float* __restrict__ scale2, float* __restrict__ bias2) {
    int t = threadIdx.x;
    if (t >= OUTP) return;
    float s = 0.f, q = 0.f;
    for (int b = 0; b < 512; ++b) { s += p2sum[b * OUTP + t]; q += p2sq[b * OUTP + t]; }
    double mu = (double)s / (double)NB;
    double var = (double)q / (double)NB - mu * mu;
    float inv = rsqrtf((float)var + BN_EPS);
    float g = (t < NOUT) ? g2[t] : 0.f;
    float bb = (t < NOUT) ? b2[t] : 0.f;
    float sc = g * inv;
    scale2[t] = sc;
    bias2[t] = bb - (float)mu * sc;
}

__global__ void k_apply(const float* __restrict__ o_raw, const float* __restrict__ scale2,
                        const float* __restrict__ bias2, float* __restrict__ out) {
    int i = blockIdx.x * blockDim.x + threadIdx.x;
    if (i >= NB * NOUT) return;
    int m = i / NOUT, n = i - m * NOUT;
    out[i] = o_raw[(size_t)m * OUTP + n] * scale2[n] + bias2[n];
}

extern "C" void kernel_launch(void* const* d_in, const int* in_sizes, int n_in,
                              void* d_out, int out_size, void* d_ws, size_t ws_size,
                              hipStream_t stream) {
    (void)in_sizes; (void)n_in; (void)out_size; (void)ws_size;
    const float* x  = (const float*)d_in[0];
    const float* W1 = (const float*)d_in[1];
    const float* W2 = (const float*)d_in[2];
    const float* g1 = (const float*)d_in[3];
    const float* b1 = (const float*)d_in[4];
    const float* g2 = (const float*)d_in[5];
    const float* b2 = (const float*)d_in[6];
    float* out = (float*)d_out;

    char* ws = (char*)d_ws;
    size_t off = 0;
    auto alloc = [&](size_t bytes) -> void* {
        void* p = ws + off;
        off = (off + bytes + 255) & ~(size_t)255;
        return p;
    };
    int8_t*  xq    = (int8_t*)alloc((size_t)NB * IND);
    int8_t*  w1q   = (int8_t*)alloc((size_t)HID * IND);
    int8_t*  w2q   = (int8_t*)alloc((size_t)OUTP * HID);
    int16_t* h16   = (int16_t*)alloc((size_t)NB * HID * 2);
    float*   o_raw = (float*)alloc((size_t)NB * OUTP * 4);
    int*     p1sum = (int*)alloc((size_t)256 * HID * 4);
    int*     p1sq  = (int*)alloc((size_t)256 * HID * 4);
    float*   sc1   = (float*)alloc(HID * 4);
    float*   bs1   = (float*)alloc(HID * 4);
    float*   p2sum = (float*)alloc((size_t)512 * OUTP * 4);
    float*   p2sq  = (float*)alloc((size_t)512 * OUTP * 4);
    float*   sc2   = (float*)alloc(OUTP * 4);
    float*   bs2   = (float*)alloc(OUTP * 4);

    k_bin4<<<(NB * IND / 4 + 255) / 256, 256, 0, stream>>>(x, xq, NB * IND / 4);
    k_bin4<<<(HID * IND / 4 + 255) / 256, 256, 0, stream>>>(W1, w1q, HID * IND / 4);
    k_bin_w2<<<(OUTP * HID + 255) / 256, 256, 0, stream>>>(W2, w2q);

    k_gemm1<<<(NB / 128) * (HID / 64), 256, 0, stream>>>(xq, w1q, h16);
    k_red1<<<256, 256, 0, stream>>>(h16, p1sum, p1sq);
    k_fin1<<<2, 256, 0, stream>>>(p1sum, p1sq, g1, b1, sc1, bs1);

    k_gemm2<<<NB / 128, 256, 0, stream>>>(h16, w2q, sc1, bs1, o_raw, p2sum, p2sq);
    k_fin2<<<1, 256, 0, stream>>>(p2sum, p2sq, g2, b2, sc2, bs2);
    k_apply<<<(NB * NOUT + 255) / 256, 256, 0, stream>>>(o_raw, sc2, bs2, out);
}
